// TriplaneDecoder_64768106824026
// MI455X (gfx1250) — matrix operand order repaired and checked by reading, compiled, not run on hardware
//
#include <hip/hip_runtime.h>
#include <hip/hip_bf16.h>
#include <math.h>

typedef _Float16 f16;
typedef __attribute__((ext_vector_type(16))) _Float16 v16h;
typedef __attribute__((ext_vector_type(8)))  _Float16 v8h;
typedef __attribute__((ext_vector_type(8)))  float    v8f;

#define WMMA_F16(A, B, C) \
  __builtin_amdgcn_wmma_f32_16x16x32_f16(false, (A), false, (B), (short)0, (C), false, false)

namespace {

struct PlaneSample {
  const float* p00; const float* p01; const float* p10; const float* p11;
  float w00, w01, w10, w11;
};

__device__ inline PlaneSample samplePlane(const float* plane, float cu, float cv) {
  float u = fminf(fmaxf((cu + 1.0f) * 0.5f * 511.0f, 0.0f), 511.0f);
  float v = fminf(fmaxf((cv + 1.0f) * 0.5f * 511.0f, 0.0f), 511.0f);
  int u0 = (int)u; if (u0 > 511) u0 = 511;
  int v0 = (int)v; if (v0 > 511) v0 = 511;
  int u1 = min(u0 + 1, 511);
  int v1 = min(v0 + 1, 511);
  float wu = u - (float)u0;
  float wv = v - (float)v0;
  PlaneSample s;
  s.p00 = plane + ((size_t)u0 * 512 + (size_t)v0) * 32;
  s.p01 = plane + ((size_t)u0 * 512 + (size_t)v1) * 32;
  s.p10 = plane + ((size_t)u1 * 512 + (size_t)v0) * 32;
  s.p11 = plane + ((size_t)u1 * 512 + (size_t)v1) * 32;
  s.w00 = (1.0f - wu) * (1.0f - wv);
  s.w01 = (1.0f - wu) * wv;
  s.w10 = wu * (1.0f - wv);
  s.w11 = wu * wv;
  return s;
}

__device__ inline void fetch4(const PlaneSample& s, int f, float o[4]) {
  float4 q00 = *(const float4*)(s.p00 + f);
  float4 q01 = *(const float4*)(s.p01 + f);
  float4 q10 = *(const float4*)(s.p10 + f);
  float4 q11 = *(const float4*)(s.p11 + f);
  o[0] = s.w00 * q00.x + s.w01 * q01.x + s.w10 * q10.x + s.w11 * q11.x;
  o[1] = s.w00 * q00.y + s.w01 * q01.y + s.w10 * q10.y + s.w11 * q11.y;
  o[2] = s.w00 * q00.z + s.w01 * q01.z + s.w10 * q10.z + s.w11 * q11.z;
  o[3] = s.w00 * q00.w + s.w01 * q01.w + s.w10 * q10.w + s.w11 * q11.w;
}

// A-fragment (16x32 f16, MxK) from row-major LDS tile.
// Contiguous k order used here: lane<16 -> M=lane,    K = kOff + {0..15}
//               lane>=16 -> M=lane-16, K = kOff + {16..31}
__device__ inline v16h ldsLoadA(const f16* tile, int stride, int lm, int hi, int kOff) {
  const f16* p = tile + lm * stride + kOff + hi * 16;
  v8h lo = *(const v8h*)(p);
  v8h hh = *(const v8h*)(p + 8);
  v16h r;
#pragma unroll
  for (int j = 0; j < 8; ++j) { r[j] = lo[j]; r[8 + j] = hh[j]; }
  return r;
}

// B-fragment (32x16 f16, KxN) from weights stored transposed [N][K] in LDS.
// Contiguous k order (matches ldsLoadA): lanes 0-15 hold K=0..15, lanes 16-31 hold K=16..31, N = lane&15
// -> 16 contiguous halves per lane.
__device__ inline v16h ldsLoadB(const f16* wt, int Kdim, int col, int kc, int hi) {
  return *(const v16h*)(wt + col * Kdim + kc * 32 + hi * 16);
}

} // namespace

__global__ __launch_bounds__(256) void triplane_decoder_kernel(
    const float* __restrict__ x_world,
    const float* __restrict__ p_zh,
    const float* __restrict__ p_hw,
    const float* __restrict__ p_zw,
    const float* __restrict__ W1,   // [32][64]
    const float* __restrict__ W2,   // [64][64]
    const float* __restrict__ W3,   // [64][4]
    float* __restrict__ out_c,      // [N][3]
    float* __restrict__ out_sigma,  // [N]
    int npts) {
  __shared__ __align__(32) f16 sFeat[256 * 32];     // 16 KB: per-point features
  __shared__ __align__(32) f16 sW1T[64 * 32];       //  4 KB: W1^T [n][k]
  __shared__ __align__(32) f16 sW2T[64 * 64];       //  8 KB: W2^T [n][k]
  __shared__ __align__(32) f16 sW3T[4 * 64];        // .5 KB: W3^T [n][k]
  __shared__ __align__(32) f16 sHid[8 * 16 * 64];   // 16 KB: per-wave hidden tile
  __shared__ unsigned char sMask[256];

  const int tid = threadIdx.x;
  const int blockBase = blockIdx.x * 256;

  // ---- Stage weights (f32 -> f16, transposed to [N][K]) ----
  for (int idx = tid; idx < 64 * 32; idx += 256) {
    int n = idx >> 5, k = idx & 31;
    sW1T[n * 32 + k] = (f16)W1[k * 64 + n];
  }
  for (int idx = tid; idx < 64 * 64; idx += 256) {
    int n = idx >> 6, k = idx & 63;
    sW2T[n * 64 + k] = (f16)W2[k * 64 + n];
  }
  for (int idx = tid; idx < 4 * 64; idx += 256) {
    int n = idx >> 6, k = idx & 63;
    sW3T[n * 64 + k] = (f16)W3[k * 4 + n];
  }

  // ---- Phase 1: bilinear gathers + feature product (one point per thread) ----
  {
    const int gp = blockBase + tid;
    const bool valid = gp < npts;
    float x0 = 0.0f, x1 = 0.0f, x2 = 0.0f;
    if (valid) {
      x0 = x_world[(size_t)gp * 3 + 0];
      x1 = x_world[(size_t)gp * 3 + 1];
      x2 = x_world[(size_t)gp * 3 + 2];
    }
    const bool inb = valid && (fabsf(x0) < 1.0f) && (fabsf(x1) < 1.0f) && (fabsf(x2) < 1.0f);
    sMask[tid] = inb ? 1 : 0;

    PlaneSample szh = samplePlane(p_zh, x0, x1);
    PlaneSample shw = samplePlane(p_hw, x1, x2);
    PlaneSample szw = samplePlane(p_zw, x2, x0);

#pragma unroll
    for (int f = 0; f < 32; f += 4) {
      float a[4], b[4], c[4];
      fetch4(szh, f, a);
      fetch4(shw, f, b);
      fetch4(szw, f, c);
#pragma unroll
      for (int j = 0; j < 4; ++j)
        sFeat[tid * 32 + f + j] = valid ? (f16)(a[j] * b[j] * c[j]) : (f16)0.0f;
    }
  }

  __syncthreads();

  // ---- Phase 2: MLP via WMMA. Each wave owns 32 points = two 16-row tiles ----
  const int wave = tid >> 5;
  const int lane = tid & 31;
  const int lm = lane & 15;   // N column / M row within half-wave
  const int hi = lane >> 4;   // half-wave select
  f16* hid = &sHid[wave * 16 * 64];

  for (int t = 0; t < 2; ++t) {
    const int tileStart = wave * 32 + t * 16;

    // Layer 1: [16x32] @ [32x64] -> [16x64], ReLU
    {
      v16h a = ldsLoadA(&sFeat[tileStart * 32], 32, lm, hi, 0);
      v8f acc[4] = {};
#pragma unroll
      for (int nt = 0; nt < 4; ++nt) {
        v16h b = ldsLoadB(sW1T, 32, nt * 16 + lm, 0, hi);
        acc[nt] = WMMA_F16(a, b, acc[nt]);
      }
#pragma unroll
      for (int nt = 0; nt < 4; ++nt)
#pragma unroll
        for (int r = 0; r < 8; ++r)
          hid[(r + hi * 8) * 64 + nt * 16 + lm] = (f16)fmaxf(acc[nt][r], 0.0f);
    }

    // Layer 2: [16x64] @ [64x64] -> [16x64], ReLU
    {
      v16h a0 = ldsLoadA(hid, 64, lm, hi, 0);
      v16h a1 = ldsLoadA(hid, 64, lm, hi, 32);
      v8f acc[4] = {};
#pragma unroll
      for (int nt = 0; nt < 4; ++nt) {
        v16h b0 = ldsLoadB(sW2T, 64, nt * 16 + lm, 0, hi);
        acc[nt] = WMMA_F16(a0, b0, acc[nt]);
        v16h b1 = ldsLoadB(sW2T, 64, nt * 16 + lm, 1, hi);
        acc[nt] = WMMA_F16(a1, b1, acc[nt]);
      }
#pragma unroll
      for (int nt = 0; nt < 4; ++nt)
#pragma unroll
        for (int r = 0; r < 8; ++r)
          hid[(r + hi * 8) * 64 + nt * 16 + lm] = (f16)fmaxf(acc[nt][r], 0.0f);
    }

    // Layer 3: [16x64] @ [64x4] -> [16x4] (N padded to 16 with zero columns)
    {
      v16h a0 = ldsLoadA(hid, 64, lm, hi, 0);
      v16h a1 = ldsLoadA(hid, 64, lm, hi, 32);
      v16h b0 = {};
      v16h b1 = {};
      if (lm < 4) {
        b0 = ldsLoadB(sW3T, 64, lm, 0, hi);
        b1 = ldsLoadB(sW3T, 64, lm, 1, hi);
      }
      v8f acc = {};
      acc = WMMA_F16(a0, b0, acc);
      acc = WMMA_F16(a1, b1, acc);

      // Epilogue: lanes with col<4 own output channel `lm` for 8 rows each.
      if (lm < 4) {
#pragma unroll
        for (int r = 0; r < 8; ++r) {
          const int p = tileStart + r + hi * 8;
          const int gp = blockBase + p;
          if (gp < npts) {
            const bool m = sMask[p] != 0;
            const float v = acc[r];
            if (lm < 3) {
              out_c[(size_t)gp * 3 + lm] = m ? (1.0f / (1.0f + expf(-v))) : 0.0f;
            } else {
              out_sigma[gp] = m ? expf(v) : 0.0f;
            }
          }
        }
      }
    }
  }
}

extern "C" void kernel_launch(void* const* d_in, const int* in_sizes, int n_in,
                              void* d_out, int out_size, void* d_ws, size_t ws_size,
                              hipStream_t stream) {
  const float* x_world = (const float*)d_in[0];
  const float* p_zh    = (const float*)d_in[1];
  const float* p_hw    = (const float*)d_in[2];
  const float* p_zw    = (const float*)d_in[3];
  const float* W1      = (const float*)d_in[4];
  const float* W2      = (const float*)d_in[5];
  const float* W3      = (const float*)d_in[6];

  const int npts = in_sizes[0] / 3;
  float* out_c     = (float*)d_out;
  float* out_sigma = out_c + (size_t)npts * 3;

  const int blocks = (npts + 255) / 256;
  triplane_decoder_kernel<<<blocks, 256, 0, stream>>>(
      x_world, p_zh, p_hw, p_zw, W1, W2, W3, out_c, out_sigma, npts);
}
